// PhenoCapsuleHead_88828513616417
// MI455X (gfx1250) — matrix operand order repaired and checked by reading, not compiled, not run on hardware
//
#include <hip/hip_runtime.h>
#include <stdint.h>

// ---------------------------------------------------------------------------
// PhenoCapsuleHead for MI455X (gfx1250, wave32, WMMA).
// GEMMs use v_wmma_f32_16x16x32_bf16 with hi/lo bf16 splitting of both
// operands (3 WMMAs per K-chunk) for near-fp32 accuracy. A-operands (z, ln(z))
// are pre-split to bf16 once (L2-resident) so the GEMM hot loop has no A-side
// conversion ALU; B-side (w1/w2 stream) packs with single v_perm_b32 ops.
// Routing recomputes votes V from L2-resident pose/w_caps instead of
// streaming a 210MB V tensor through HBM 6 times.
// ---------------------------------------------------------------------------

#define D_IN   1024
#define N_PRI  2048
#define D_PRI  16
#define N_CLS  25
#define D_CLS  16
#define OD     400        // N_CLS * D_CLS
#define BATCH  64
#define EPSR   1e-6f
#define NB     4          // batch rows per routing block
#define NCH    16         // n-chunks (128 n's each)

typedef float    v8f   __attribute__((ext_vector_type(8)));
typedef __bf16   v16bf __attribute__((ext_vector_type(16)));
typedef uint32_t v8u   __attribute__((ext_vector_type(8)));

struct bf16x16 { uint4 a, b; };   // 32 bytes -> v16bf

__device__ __forceinline__ float hif(float f) {
  return __uint_as_float(__float_as_uint(f) & 0xFFFF0000u);
}
// pack truncated-bf16(f0) into low half, truncated-bf16(f1) into high half:
// single v_perm_b32 selecting the top bytes of each dword.
__device__ __forceinline__ uint32_t pk_perm(float f0, float f1) {
  return __builtin_amdgcn_perm(__float_as_uint(f1), __float_as_uint(f0), 0x07060302u);
}

// ---------------------------------------------------------------------------
// Split a fp32 array into truncated-bf16 hi and residual-bf16 lo arrays.
// ---------------------------------------------------------------------------
__global__ __launch_bounds__(256)
void k_cvt(const float* __restrict__ in, uint16_t* __restrict__ hi,
           uint16_t* __restrict__ lo, int n) {
  const int i = blockIdx.x * 256 + threadIdx.x;
  if (i >= n) return;
  const float f = in[i];
  const uint32_t u = __float_as_uint(f);
  hi[i] = (uint16_t)(u >> 16);
  const float l = f - __uint_as_float(u & 0xFFFF0000u);
  lo[i] = (uint16_t)(__float_as_uint(l) >> 16);
}

// ---------------------------------------------------------------------------
// LayerNorm over D_IN for each of 64 rows.
// ---------------------------------------------------------------------------
__global__ __launch_bounds__(256)
void k_ln(const float* __restrict__ z, const float* __restrict__ g,
          const float* __restrict__ b, float* __restrict__ out) {
  __shared__ float red[256];
  const int row = blockIdx.x, t = threadIdx.x;
  const float* zr = z + (size_t)row * D_IN;
  float s = 0.f;
  for (int j = t; j < D_IN; j += 256) s += zr[j];
  red[t] = s; __syncthreads();
  for (int k = 128; k; k >>= 1) { if (t < k) red[t] += red[t + k]; __syncthreads(); }
  const float mean = red[0] * (1.f / D_IN);
  __syncthreads();
  float vs = 0.f;
  for (int j = t; j < D_IN; j += 256) { float d = zr[j] - mean; vs += d * d; }
  red[t] = vs; __syncthreads();
  for (int k = 128; k; k >>= 1) { if (t < k) red[t] += red[t + k]; __syncthreads(); }
  const float inv = rsqrtf(red[0] * (1.f / D_IN) + 1e-5f);
  for (int j = t; j < D_IN; j += 256)
    out[(size_t)row * D_IN + j] = (zr[j] - mean) * inv * g[j] + b[j];
}

// ---------------------------------------------------------------------------
// WMMA GEMM: C(64 x N) = A(64 x 1024) @ B(1024 x N) + bias, row-major.
// A supplied pre-split as bf16 hi/lo (loaded straight into WMMA operands);
// B (the HBM stream) converted inline with v_perm_b32 packing.
// One wave owns one 16-col N tile and all four 16-row M tiles, so each B
// element is loaded from HBM exactly once. EPI=1 applies sigmoid.
// ---------------------------------------------------------------------------
template <int EPI>
__global__ __launch_bounds__(256)
void k_gemm_bf16(const uint16_t* __restrict__ Ahi, const uint16_t* __restrict__ Alo,
                 const float* __restrict__ Bm, const float* __restrict__ bias,
                 float* __restrict__ C, int N) {
  const int lane = threadIdx.x & 31;
  const int wv   = threadIdx.x >> 5;
  const int n0   = (blockIdx.x * 8 + wv) * 16;
  const int col  = lane & 15;
  const int half = lane >> 4;
  const int nc   = n0 + col;

  v8f acc[4];
#pragma unroll
  for (int mt = 0; mt < 4; ++mt)
#pragma unroll
    for (int v = 0; v < 8; ++v) acc[mt][v] = 0.f;

  for (int k0 = 0; k0 < D_IN; k0 += 32) {
    // --- B tile (32 x 16): lanes 0-15 hold K=k0..k0+15, lanes 16-31 K=+16..+31
    const int kB = k0 + half * 16;
    // prefetch next chunk of the strided w-stream
    __builtin_prefetch(&Bm[(size_t)(kB + 32) * N + nc], 0, 0);
    v8u bhu, blu;
#pragma unroll
    for (int h = 0; h < 8; ++h) {
      const float f0 = Bm[(size_t)(kB + 2 * h) * N + nc];
      const float f1 = Bm[(size_t)(kB + 2 * h + 1) * N + nc];
      bhu[h] = pk_perm(f0, f1);
      blu[h] = pk_perm(f0 - hif(f0), f1 - hif(f1));
    }
    const v16bf bh = __builtin_bit_cast(v16bf, bhu);
    const v16bf bl = __builtin_bit_cast(v16bf, blu);

    // --- A tiles (16 x 32): K runs [kA,kA+16) per lane half
    const int kA = k0 + half * 16;
#pragma unroll
    for (int mt = 0; mt < 4; ++mt) {
      const size_t base = (size_t)(mt * 16 + col) * D_IN + kA;
      bf16x16 ah, al;
      ah.a = *(const uint4*)(Ahi + base);
      ah.b = *(const uint4*)(Ahi + base + 8);
      al.a = *(const uint4*)(Alo + base);
      al.b = *(const uint4*)(Alo + base + 8);
      const v16bf ahv = __builtin_bit_cast(v16bf, ah);
      const v16bf alv = __builtin_bit_cast(v16bf, al);
      acc[mt] = __builtin_amdgcn_wmma_f32_16x16x32_bf16(false, ahv, false, bh,
                                                        (short)0, acc[mt], false, false);
      acc[mt] = __builtin_amdgcn_wmma_f32_16x16x32_bf16(false, ahv, false, bl,
                                                        (short)0, acc[mt], false, false);
      acc[mt] = __builtin_amdgcn_wmma_f32_16x16x32_bf16(false, alv, false, bh,
                                                        (short)0, acc[mt], false, false);
    }
  }

  const float bs = bias[nc];
#pragma unroll
  for (int mt = 0; mt < 4; ++mt)
#pragma unroll
    for (int v = 0; v < 8; ++v) {
      float r = acc[mt][v] + bs;
      if (EPI == 1) r = 1.f / (1.f + expf(-r));
      C[(size_t)(mt * 16 + half * 8 + v) * N + nc] = r;
    }
}

// ---------------------------------------------------------------------------
// M-step: recompute votes V[b,n,o,d] = sum_i pose[b,n,i]*w_caps[n,i,o,d] and
// accumulate S1 = sum_n r*V, S2 = sum_n r*V^2, rsum = sum_n r into per-chunk
// partial buffers (deterministic; reduced in k_stats).
// Block: (b-group of NB, n-chunk of 128). Thread t handles (o,d)=t and t+256.
// ---------------------------------------------------------------------------
__global__ __launch_bounds__(256)
void k_mstep(const float* __restrict__ pose, const float* __restrict__ wcaps,
             const float* __restrict__ act, const float* __restrict__ R,
             float* __restrict__ P1, float* __restrict__ P2, float* __restrict__ PR,
             int uniformR) {
  __shared__ float s_pose[NB][16];
  __shared__ float s_r[NB][N_CLS];
  const int tid = threadIdx.x;
  const int b0  = blockIdx.x * NB;
  const int nbeg = blockIdx.y * (N_PRI / NCH);

  float a1[2][NB], a2[2][NB], ra[2][NB];
#pragma unroll
  for (int p = 0; p < 2; ++p)
#pragma unroll
    for (int bl = 0; bl < NB; ++bl) { a1[p][bl] = 0.f; a2[p][bl] = 0.f; ra[p][bl] = 0.f; }

  for (int n = nbeg; n < nbeg + (N_PRI / NCH); ++n) {
    if (tid < NB * 16) {
      int bl = tid >> 4, i = tid & 15;
      s_pose[bl][i] = pose[((size_t)(b0 + bl) * N_PRI + n) * D_PRI + i];
    }
    if (tid < NB * N_CLS) {
      int bl = tid / N_CLS, o = tid - bl * N_CLS;
      float av = act[(size_t)(b0 + bl) * N_PRI + n];
      s_r[bl][o] = uniformR ? av * (1.f / N_CLS)
                            : R[((size_t)(b0 + bl) * N_PRI + n) * N_CLS + o] * av;
    }
    __syncthreads();
#pragma unroll
    for (int p = 0; p < 2; ++p) {
      const int t = tid + p * 256;
      if (t < OD) {
        const int o = t >> 4, d = t & 15;
        const float* wp = wcaps + (size_t)n * (D_PRI * OD) + t;
        float v[NB];
#pragma unroll
        for (int bl = 0; bl < NB; ++bl) v[bl] = 0.f;
#pragma unroll
        for (int i = 0; i < D_PRI; ++i) {
          const float w = wp[(size_t)i * OD];
#pragma unroll
          for (int bl = 0; bl < NB; ++bl) v[bl] = fmaf(s_pose[bl][i], w, v[bl]);
        }
#pragma unroll
        for (int bl = 0; bl < NB; ++bl) {
          const float r = s_r[bl][o], vb = v[bl];
          a1[p][bl] += r * vb;
          a2[p][bl] += r * vb * vb;
          if (d == 0) ra[p][bl] += r;
        }
      }
    }
    __syncthreads();
  }

  // flush partials for this n-chunk
#pragma unroll
  for (int p = 0; p < 2; ++p) {
    const int t = tid + p * 256;
    if (t < OD) {
#pragma unroll
      for (int bl = 0; bl < NB; ++bl) {
        const size_t pb = (size_t)blockIdx.y * BATCH + (b0 + bl);
        P1[pb * OD + t] = a1[p][bl];
        P2[pb * OD + t] = a2[p][bl];
        if ((t & 15) == 0) PR[pb * N_CLS + (t >> 4)] = ra[p][bl];
      }
    }
  }
}

// ---------------------------------------------------------------------------
// Stats: reduce chunk partials, compute mu, sigma^2, a_out and E-step
// constants: muinv = mu/sig2, hinv = 0.5/sig2,
// c0 = log(a+eps) - 0.5*sum(log sig2) - 0.5*sum(mu^2/sig2).
// One thread per (b,o) pair (1600 total).
// ---------------------------------------------------------------------------
__global__ __launch_bounds__(256)
void k_stats(const float* __restrict__ P1, const float* __restrict__ P2,
             const float* __restrict__ PR, const float* __restrict__ bu,
             const float* __restrict__ ba, float* __restrict__ mu,
             float* __restrict__ muinv, float* __restrict__ hinv,
             float* __restrict__ c0) {
  const int idx = blockIdx.x * 256 + threadIdx.x;
  if (idx >= BATCH * N_CLS) return;
  const int b = idx / N_CLS, o = idx - b * N_CLS;

  float rs_raw = 0.f;
#pragma unroll
  for (int c = 0; c < NCH; ++c) rs_raw += PR[((size_t)c * BATCH + b) * N_CLS + o];
  const float rs = rs_raw + EPSR;
  const float inv_rs = 1.f / rs;

  float logsum = 0.f, musq = 0.f;
  float muv[D_CLS], sg[D_CLS];
#pragma unroll
  for (int d = 0; d < D_CLS; ++d) {
    float s1 = 0.f, s2 = 0.f;
#pragma unroll
    for (int c = 0; c < NCH; ++c) {
      const size_t pb = ((size_t)c * BATCH + b) * OD + o * D_CLS + d;
      s1 += P1[pb]; s2 += P2[pb];
    }
    const float m  = s1 * inv_rs;
    const float v2 = (s2 - 2.f * m * s1 + m * m * rs_raw) * inv_rs + EPSR;
    muv[d] = m; sg[d] = v2;
    logsum += logf(v2);
    musq   += m * m / v2;
  }
  const float cost = (16.f * bu[o] + 0.5f * logsum) * rs;
  const float a = 1.f / (1.f + expf(-(ba[o] - cost)));
  c0[idx] = logf(a + EPSR) - 0.5f * logsum - 0.5f * musq;
#pragma unroll
  for (int d = 0; d < D_CLS; ++d) {
    mu[(size_t)idx * D_CLS + d]    = muv[d];
    muinv[(size_t)idx * D_CLS + d] = muv[d] / sg[d];
    hinv[(size_t)idx * D_CLS + d]  = 0.5f / sg[d];
  }
}

// ---------------------------------------------------------------------------
// E-step: recompute V, logit[b,n,o] = c0[b,o] + sum_d (V*muinv - V^2*hinv),
// R = softmax over o. Writes R (last iteration writes q directly to d_out).
// ---------------------------------------------------------------------------
__global__ __launch_bounds__(256)
void k_estep(const float* __restrict__ pose, const float* __restrict__ wcaps,
             const float* __restrict__ muinv, const float* __restrict__ hinv,
             const float* __restrict__ c0, float* __restrict__ Rout) {
  __shared__ float s_pose[NB][16];
  __shared__ float s_mi[NB * OD];
  __shared__ float s_h2[NB * OD];
  __shared__ float s_c0[NB * N_CLS];
  __shared__ float s_p[NB][OD];
  __shared__ float s_l[NB * N_CLS];
  __shared__ float s_mx[NB], s_si[NB];

  const int tid = threadIdx.x;
  const int b0  = blockIdx.x * NB;
  const int nbeg = blockIdx.y * (N_PRI / NCH);

  for (int j = tid; j < NB * OD; j += 256) {
    s_mi[j] = muinv[(size_t)b0 * OD + j];
    s_h2[j] = hinv[(size_t)b0 * OD + j];
  }
  if (tid < NB * N_CLS) s_c0[tid] = c0[(size_t)b0 * N_CLS + tid];
  __syncthreads();

  for (int n = nbeg; n < nbeg + (N_PRI / NCH); ++n) {
    if (tid < NB * 16) {
      int bl = tid >> 4, i = tid & 15;
      s_pose[bl][i] = pose[((size_t)(b0 + bl) * N_PRI + n) * D_PRI + i];
    }
    __syncthreads();
#pragma unroll
    for (int p = 0; p < 2; ++p) {
      const int t = tid + p * 256;
      if (t < OD) {
        const float* wp = wcaps + (size_t)n * (D_PRI * OD) + t;
        float v[NB];
#pragma unroll
        for (int bl = 0; bl < NB; ++bl) v[bl] = 0.f;
#pragma unroll
        for (int i = 0; i < D_PRI; ++i) {
          const float w = wp[(size_t)i * OD];
#pragma unroll
          for (int bl = 0; bl < NB; ++bl) v[bl] = fmaf(s_pose[bl][i], w, v[bl]);
        }
#pragma unroll
        for (int bl = 0; bl < NB; ++bl) {
          const float vb = v[bl];
          s_p[bl][t] = vb * s_mi[bl * OD + t] - vb * vb * s_h2[bl * OD + t];
        }
      }
    }
    __syncthreads();
    if (tid < NB * N_CLS) {
      const int bl = tid / N_CLS, o = tid - bl * N_CLS;
      float s = 0.f;
#pragma unroll
      for (int d = 0; d < D_CLS; ++d) s += s_p[bl][o * D_CLS + d];
      s_l[tid] = s_c0[tid] + s;
    }
    __syncthreads();
    if (tid < NB) {
      float mx = -3.402823e38f;
      for (int o = 0; o < N_CLS; ++o) mx = fmaxf(mx, s_l[tid * N_CLS + o]);
      float ss = 0.f;
      for (int o = 0; o < N_CLS; ++o) ss += expf(s_l[tid * N_CLS + o] - mx);
      s_mx[tid] = mx; s_si[tid] = 1.f / ss;
    }
    __syncthreads();
    if (tid < NB * N_CLS) {
      const int bl = tid / N_CLS, o = tid - bl * N_CLS;
      Rout[((size_t)(b0 + bl) * N_PRI + n) * N_CLS + o] =
          expf(s_l[tid] - s_mx[bl]) * s_si[bl];
    }
    __syncthreads();
  }
}

// ---------------------------------------------------------------------------
// Final head: LN over d=16, logits, pose norms, copy class_pose.
// ---------------------------------------------------------------------------
__global__ __launch_bounds__(256)
void k_final(const float* __restrict__ mu, const float* __restrict__ g,
             const float* __restrict__ bb, const float* __restrict__ w3,
             const float* __restrict__ b3, float* __restrict__ out) {
  const int idx = blockIdx.x * 256 + threadIdx.x;
  if (idx >= BATCH * N_CLS) return;
  const float* m = mu + (size_t)idx * D_CLS;
  float mean = 0.f, nsq = 0.f;
#pragma unroll
  for (int d = 0; d < D_CLS; ++d) { mean += m[d]; nsq += m[d] * m[d]; }
  mean *= (1.f / D_CLS);
  float var = 0.f;
#pragma unroll
  for (int d = 0; d < D_CLS; ++d) { float dd = m[d] - mean; var += dd * dd; }
  var *= (1.f / D_CLS);
  const float inv = rsqrtf(var + 1e-5f);
  float lg = 0.f;
#pragma unroll
  for (int d = 0; d < D_CLS; ++d)
    lg += ((m[d] - mean) * inv * g[d] + bb[d]) * w3[d];
  lg += b3[0];
  out[idx]        = lg;            // logits
  out[1600 + idx] = sqrtf(nsq);    // acts = ||mu||
#pragma unroll
  for (int d = 0; d < D_CLS; ++d)
    out[3200 + (size_t)idx * D_CLS + d] = m[d];  // class_pose
}

// ---------------------------------------------------------------------------
extern "C" void kernel_launch(void* const* d_in, const int* in_sizes, int n_in,
                              void* d_out, int out_size, void* d_ws, size_t ws_size,
                              hipStream_t stream) {
  (void)in_sizes; (void)n_in; (void)out_size; (void)ws_size;
  const float* z   = (const float*)d_in[0];
  const float* w1  = (const float*)d_in[1];
  const float* b1  = (const float*)d_in[2];
  const float* g1  = (const float*)d_in[3];
  const float* bb1 = (const float*)d_in[4];
  const float* w2  = (const float*)d_in[5];
  const float* b2  = (const float*)d_in[6];
  const float* wc  = (const float*)d_in[7];
  const float* bu  = (const float*)d_in[8];
  const float* ba  = (const float*)d_in[9];
  const float* g2  = (const float*)d_in[10];
  const float* bb2 = (const float*)d_in[11];
  const float* w3  = (const float*)d_in[12];
  const float* b3  = (const float*)d_in[13];
  float* out = (float*)d_out;

  float* ws    = (float*)d_ws;
  float* zln   = ws;                         // 65536
  float* pose  = zln + 65536;                // 2097152
  float* act   = pose + 2097152;             // 131072
  float* Rw    = act + 131072;               // 3276800
  float* P1    = Rw + 3276800;               // 16*64*400 = 409600
  float* P2    = P1 + 409600;                // 409600
  float* PR    = P2 + 409600;                // 16*64*25 = 25600
  float* mu    = PR + 25600;                 // 25600
  float* muinv = mu + 25600;                 // 25600
  float* hinv  = muinv + 25600;              // 25600
  float* c0    = hinv + 25600;               // 1600
  uint16_t* zhi   = (uint16_t*)(c0 + 1600);  // 65536 u16
  uint16_t* zlo   = zhi + 65536;
  uint16_t* zlnhi = zlo + 65536;
  uint16_t* zlnlo = zlnhi + 65536;

  k_ln<<<BATCH, 256, 0, stream>>>(z, g1, bb1, zln);
  k_cvt<<<256, 256, 0, stream>>>(z,   zhi,   zlo,   BATCH * D_IN);
  k_cvt<<<256, 256, 0, stream>>>(zln, zlnhi, zlnlo, BATCH * D_IN);

  // pose = z @ w1 + b1   (N = 32768 -> 2048 tiles / 8 waves = 256 blocks)
  k_gemm_bf16<0><<<256, 256, 0, stream>>>(zhi, zlo, w1, b1, pose, N_PRI * D_PRI);
  // act = sigmoid(ln(z) @ w2 + b2)  (N = 2048 -> 16 blocks)
  k_gemm_bf16<1><<<16, 256, 0, stream>>>(zlnhi, zlnlo, w2, b2, act, N_PRI);

  for (int it = 0; it < 3; ++it) {
    k_mstep<<<dim3(BATCH / NB, NCH), 256, 0, stream>>>(pose, wc, act, Rw,
                                                       P1, P2, PR, it == 0 ? 1 : 0);
    k_stats<<<7, 256, 0, stream>>>(P1, P2, PR, bu, ba, mu, muinv, hinv, c0);
    float* Rout = (it == 2) ? (out + 28800) : Rw;   // last E-step -> q output
    k_estep<<<dim3(BATCH / NB, NCH), 256, 0, stream>>>(pose, wc, muinv, hinv, c0, Rout);
  }
  k_final<<<7, 256, 0, stream>>>(mu, g2, bb2, w3, b3, out);
}
